// Mamba3_64819646431743
// MI455X (gfx1250) — matrix operand order repaired and checked by reading, not compiled, not run on hardware
//
#include <hip/hip_runtime.h>
#include <hip/hip_bf16.h>

// ---------------- problem constants ----------------
#define DMODEL   1024
#define DSTATE   64
#define DCONV    4
#define HEADDIM  128
#define NHEADS   16
#define DINNER   2048
#define DINPROJ  4240           // 2*DINNER + 2*DSTATE + NHEADS
#define DINPROJ_PAD 4288        // DINPROJ rounded up to 64
#define CONVDIM  2176           // DINNER + 2*DSTATE
#define BATCH    2
#define SEQ      1024
#define ROWS     (BATCH * SEQ)  // 2048 tokens

typedef __attribute__((ext_vector_type(16))) __bf16          bf16x16;
typedef __attribute__((ext_vector_type(8)))  float           f32x8;
typedef __attribute__((ext_vector_type(8)))  unsigned short  u16x8;

struct BFrag { union { unsigned short u[16]; u16x8 h[2]; bf16x16 v; }; };

// round-to-nearest-even fp32 -> bf16 (used in the streaming converter and gate_norm_kernel)
__device__ __forceinline__ unsigned short f2bf(float f) {
  unsigned int x = __builtin_bit_cast(unsigned int, f);
  x += 0x7FFFu + ((x >> 16) & 1u);
  return (unsigned short)(x >> 16);
}

// =====================================================================
// Streaming fp32 -> bf16 converter with row padding: rows [rows,rowsPad)
// are zero-filled so the GEMM B-side never needs edge masking.
// cols must be a multiple of 4 (1024 / 2048 here).
// =====================================================================
__global__ void f32_to_bf16_pad_kernel(const float* __restrict__ src,
                                       unsigned short* __restrict__ dst,
                                       int rows, int cols, int rowsPad)
{
  int i = (blockIdx.x * blockDim.x + threadIdx.x) * 4;
  if (i >= rowsPad * cols) return;
  if (i / cols < rows) {
    float4 v = *(const float4*)(src + i);
    dst[i + 0] = f2bf(v.x);
    dst[i + 1] = f2bf(v.y);
    dst[i + 2] = f2bf(v.z);
    dst[i + 3] = f2bf(v.w);
  } else {
    dst[i + 0] = 0; dst[i + 1] = 0; dst[i + 2] = 0; dst[i + 3] = 0;
  }
}

// =====================================================================
// WMMA GEMM:  C[M,N](f32) = A[M,K](bf16) * Bm[Npad,K](bf16)^T
// Compile-time M/N/K: fragment strides mi*16*K become immediate load
// offsets, so the K-loop is just {2 base increments, 16 b128 loads,
// 16 v_wmma_f32_16x16x32_bf16} — no edge masks, no address VALU chains.
// One wave owns a 64x64 output tile. Bm must be padded to NSTRIPS*64
// rows (zero-filled); stores are guarded against logical N.
// =====================================================================
template <int M, int N, int K>
__global__ __launch_bounds__(256) void wmma_gemm_bf16_kernel(
    const unsigned short* __restrict__ A, const unsigned short* __restrict__ Bm,
    float* __restrict__ C)
{
  constexpr int NSTRIPS = (N + 63) / 64;
  constexpr int MSTRIPS = M / 64;

  const int lane = threadIdx.x & 31;
  const int wave = (blockIdx.x * blockDim.x + threadIdx.x) >> 5;
  if (wave >= MSTRIPS * NSTRIPS) return;

  const int ms = wave / NSTRIPS, ns = wave % NSTRIPS;
  const int m0 = ms << 6, n0 = ns << 6;
  const int lhalf = lane >> 4;        // 0 | 1
  const int lrow  = lane & 15;

  // A frag: lane half picks K sub-runs {+0..7|+8..15} and {+16..23|+24..31}
  const unsigned short* abase = A + (size_t)(m0 + lrow) * K + lhalf * 8;
  // B frag: lane = column, same K sub-runs as A {+0..7|+8..15} and {+16..23|+24..31}
  const unsigned short* bbase = Bm + (size_t)(n0 + lrow) * K + lhalf * 8;

  const f32x8 vzero = {0.f, 0.f, 0.f, 0.f, 0.f, 0.f, 0.f, 0.f};
  f32x8 acc[4][4];
#pragma unroll
  for (int mi = 0; mi < 4; ++mi)
#pragma unroll
    for (int nj = 0; nj < 4; ++nj) acc[mi][nj] = vzero;

  for (int kb = 0; kb < K; kb += 32) {
    BFrag fa[4], fb[4];
#pragma unroll
    for (int mi = 0; mi < 4; ++mi) {        // strides are compile-time
      fa[mi].h[0] = *(const u16x8*)(abase + kb + mi * 16 * K);
      fa[mi].h[1] = *(const u16x8*)(abase + kb + mi * 16 * K + 16);
    }
#pragma unroll
    for (int nj = 0; nj < 4; ++nj) {
      fb[nj].h[0] = *(const u16x8*)(bbase + kb + nj * 16 * K);
      fb[nj].h[1] = *(const u16x8*)(bbase + kb + nj * 16 * K + 16);
    }
    __builtin_prefetch(abase + kb + 128, 0, 1);   // global_prefetch_b8
    __builtin_prefetch(bbase + kb + 128, 0, 1);

#pragma unroll
    for (int mi = 0; mi < 4; ++mi)
#pragma unroll
      for (int nj = 0; nj < 4; ++nj)
        acc[mi][nj] = __builtin_amdgcn_wmma_f32_16x16x32_bf16(
            false, fa[mi].v, false, fb[nj].v,
            (short)0, acc[mi][nj], false, false);
  }

  // D layout: VGPR r -> M = r + 8*lhalf, N = lane&15
#pragma unroll
  for (int nj = 0; nj < 4; ++nj) {
    const int col = n0 + nj * 16 + lrow;
    if ((N % 64 == 0) || (col < N)) {
#pragma unroll
      for (int mi = 0; mi < 4; ++mi) {
        float* cp = C + (size_t)(m0 + mi * 16 + lhalf * 8) * N + col;
#pragma unroll
        for (int r = 0; r < 8; ++r) cp[(size_t)r * N] = acc[mi][nj][r];
      }
    }
  }
}

// =====================================================================
// Depthwise causal conv (k=4) + bias + SiLU over the xBC slice
// =====================================================================
__global__ void conv_silu_kernel(const float* __restrict__ zxbcdt,
                                 const float* __restrict__ conv_w,
                                 const float* __restrict__ conv_b,
                                 float* __restrict__ xBC)
{
  int idx = blockIdx.x * blockDim.x + threadIdx.x;
  const int total = ROWS * CONVDIM;
  if (idx >= total) return;
  int c   = idx % CONVDIM;
  int row = idx / CONVDIM;      // b*SEQ + l
  int l   = row % SEQ;
  float acc = conv_b[c];
#pragma unroll
  for (int k = 0; k < DCONV; ++k) {
    int ls = l - (DCONV - 1) + k;
    if (ls >= 0) {
      float xv = zxbcdt[(size_t)(row - (l - ls)) * DINPROJ + DINNER + c];
      acc += xv * conv_w[c * DCONV + k];
    }
  }
  float sg = 1.0f / (1.0f + __expf(-acc));
  xBC[(size_t)row * CONVDIM + c] = acc * sg;
}

// =====================================================================
// dt = softplus(dt_raw + dt_bias)
// =====================================================================
__global__ void dt_softplus_kernel(const float* __restrict__ zxbcdt,
                                   const float* __restrict__ dt_bias,
                                   float* __restrict__ dt_out)
{
  int idx = blockIdx.x * blockDim.x + threadIdx.x;
  if (idx >= ROWS * NHEADS) return;
  int h   = idx % NHEADS;
  int row = idx / NHEADS;
  float x = zxbcdt[(size_t)row * DINPROJ + DINNER + CONVDIM + h] + dt_bias[h];
  dt_out[idx] = (x > 20.f) ? x : log1pf(__expf(x));
}

// =====================================================================
// Selective scan: one workgroup per (b, head). State 128x64 lives in
// registers (32 f32 per lane). Per step, B/C/x staged via LDS; the
// C-contraction finishes with a wave32 __shfl_xor pair reduction.
// Also folds the skip term  + D_param * x.
// =====================================================================
__global__ __launch_bounds__(256) void scan_kernel(
    const float* __restrict__ xBC, const float* __restrict__ dt_soft,
    const float* __restrict__ A_log, const float* __restrict__ Dprm,
    float* __restrict__ y)
{
  const int b = blockIdx.x / NHEADS;
  const int h = blockIdx.x % NHEADS;
  const int t = threadIdx.x;
  const int p  = t >> 1;                     // head-dim row 0..127
  const int nb = (t & 1) * (DSTATE / 2);     // state half 0..31 / 32..63

  __shared__ float sB[DSTATE], sC[DSTATE], sx[HEADDIM];
  __shared__ float sdt;

  const float Aval = -__expf(A_log[h]);
  const float Dv   = Dprm[h];

  float hst[DSTATE / 2];
#pragma unroll
  for (int i = 0; i < DSTATE / 2; ++i) hst[i] = 0.f;

  for (int l = 0; l < SEQ; ++l) {
    const size_t base = (size_t)(b * SEQ + l) * CONVDIM;
    if (t < DSTATE)            sB[t]            = xBC[base + DINNER + t];
    else if (t < 2 * DSTATE)   sC[t - DSTATE]   = xBC[base + DINNER + DSTATE + (t - DSTATE)];
    else                       sx[t - 2*DSTATE] = xBC[base + h * HEADDIM + (t - 2*DSTATE)];
    if (t == 0) sdt = dt_soft[(size_t)(b * SEQ + l) * NHEADS + h];
    __syncthreads();

    const float dtv   = sdt;
    const float dA    = __expf(dtv * Aval);
    const float xv    = sx[p];
    const float coeff = dtv * xv;
    float partial = 0.f;
#pragma unroll
    for (int i = 0; i < DSTATE / 2; ++i) {
      const int n = nb + i;
      hst[i] = dA * hst[i] + coeff * sB[n];
      partial += hst[i] * sC[n];
    }
    partial += __shfl_xor(partial, 1);       // pair lanes share p
    if ((t & 1) == 0)
      y[(size_t)(b * SEQ + l) * DINNER + h * HEADDIM + p] = partial + Dv * xv;
    __syncthreads();
  }
}

// =====================================================================
// y = y * silu(z); RMSNorm over DINNER; * norm_w. One block per token.
// Emits bf16 directly so out_proj GEMM loads need no conversion.
// =====================================================================
__global__ __launch_bounds__(256) void gate_norm_kernel(
    const float* __restrict__ zxbcdt, const float* __restrict__ y_in,
    const float* __restrict__ norm_w, unsigned short* __restrict__ y_out)
{
  const int row = blockIdx.x;
  const int t   = threadIdx.x;
  float vals[DINNER / 256];
  float ss = 0.f;
#pragma unroll
  for (int i = 0; i < DINNER / 256; ++i) {
    const int e = t + i * 256;
    const float z = zxbcdt[(size_t)row * DINPROJ + e];
    const float g = z / (1.0f + __expf(-z));
    const float v = y_in[(size_t)row * DINNER + e] * g;
    vals[i] = v;
    ss += v * v;
  }
#pragma unroll
  for (int off = 16; off > 0; off >>= 1) ss += __shfl_down(ss, off);
  __shared__ float red[8];
  const int wid = t >> 5, lane = t & 31;
  if (lane == 0) red[wid] = ss;
  __syncthreads();
  if (t == 0) { float s = 0.f; for (int i = 0; i < 8; ++i) s += red[i]; red[0] = s; }
  __syncthreads();
  const float scale = rsqrtf(red[0] / (float)DINNER + 1e-5f);
#pragma unroll
  for (int i = 0; i < DINNER / 256; ++i) {
    const int e = t + i * 256;
    y_out[(size_t)row * DINNER + e] = f2bf(vals[i] * scale * norm_w[e]);
  }
}

// =====================================================================
extern "C" void kernel_launch(void* const* d_in, const int* in_sizes, int n_in,
                              void* d_out, int out_size, void* d_ws, size_t ws_size,
                              hipStream_t stream) {
  (void)in_sizes; (void)n_in; (void)out_size; (void)ws_size;
  const float* hidden  = (const float*)d_in[0];
  const float* W_in    = (const float*)d_in[1];
  const float* conv_w  = (const float*)d_in[2];
  const float* conv_b  = (const float*)d_in[3];
  const float* dt_bias = (const float*)d_in[4];
  const float* A_log   = (const float*)d_in[5];
  const float* D_param = (const float*)d_in[6];
  const float* norm_w  = (const float*)d_in[7];
  const float* W_out   = (const float*)d_in[8];
  float* out = (float*)d_out;

  // ---- workspace carve-out ----
  float* zx  = (float*)d_ws;                        // ROWS x DINPROJ   (f32)
  float* xbc = zx  + (size_t)ROWS * DINPROJ;        // ROWS x CONVDIM   (f32)
  float* dts = xbc + (size_t)ROWS * CONVDIM;        // ROWS x NHEADS    (f32)
  float* yb  = dts + (size_t)ROWS * NHEADS;         // ROWS x DINNER    (f32)
  unsigned short* hid_bf  = (unsigned short*)(yb + (size_t)ROWS * DINNER);
  unsigned short* win_bf  = hid_bf + (size_t)ROWS * DMODEL;        // DINPROJ_PAD x DMODEL
  unsigned short* wout_bf = win_bf + (size_t)DINPROJ_PAD * DMODEL; // DMODEL x DINNER
  unsigned short* yn_bf   = wout_bf + (size_t)DMODEL * DINNER;     // ROWS x DINNER

  // ---- 0) pre-convert GEMM operands to bf16 (weights row-padded) ----
  {
    int n1 = ROWS * DMODEL;
    f32_to_bf16_pad_kernel<<<(n1 / 4 + 255) / 256, 256, 0, stream>>>(
        hidden, hid_bf, ROWS, DMODEL, ROWS);
    int n2 = DINPROJ_PAD * DMODEL;
    f32_to_bf16_pad_kernel<<<(n2 / 4 + 255) / 256, 256, 0, stream>>>(
        W_in, win_bf, DINPROJ, DMODEL, DINPROJ_PAD);
    int n3 = DMODEL * DINNER;
    f32_to_bf16_pad_kernel<<<(n3 / 4 + 255) / 256, 256, 0, stream>>>(
        W_out, wout_bf, DMODEL, DINNER, DMODEL);
  }

  // ---- 1) in_proj: zx = hidden(2048x1024) * W_in(4240x1024)^T ----
  {
    constexpr int waves  = (ROWS / 64) * ((DINPROJ + 63) / 64);
    constexpr int blocks = (waves * 32 + 255) / 256;
    wmma_gemm_bf16_kernel<ROWS, DINPROJ, DMODEL>
        <<<blocks, 256, 0, stream>>>(hid_bf, win_bf, zx);
  }

  // ---- 2) depthwise conv + SiLU; dt softplus ----
  {
    const int total = ROWS * CONVDIM;
    conv_silu_kernel<<<(total + 255) / 256, 256, 0, stream>>>(zx, conv_w, conv_b, xbc);
    const int tdt = ROWS * NHEADS;
    dt_softplus_kernel<<<(tdt + 255) / 256, 256, 0, stream>>>(zx, dt_bias, dts);
  }

  // ---- 3) selective scan (+ D*x skip) — one WG per (batch, head) ----
  scan_kernel<<<BATCH * NHEADS, 256, 0, stream>>>(xbc, dts, A_log, D_param, yb);

  // ---- 4) SiLU gate + RMSNorm (emits bf16) ----
  gate_norm_kernel<<<ROWS, 256, 0, stream>>>(zx, yb, norm_w, yn_bf);

  // ---- 5) out_proj: out = yn(2048x2048) * W_out(1024x2048)^T ----
  {
    constexpr int waves  = (ROWS / 64) * (DMODEL / 64);
    constexpr int blocks = (waves * 32 + 255) / 256;
    wmma_gemm_bf16_kernel<ROWS, DMODEL, DINNER>
        <<<blocks, 256, 0, stream>>>(yn_bf, wout_bf, out);
  }
}
